// Layer_49795850830100
// MI455X (gfx1250) — matrix operand order repaired and checked by reading, not compiled, not run on hardware
//
#include <hip/hip_runtime.h>
#include <math.h>

#define BB    4
#define LL    1024
#define DD    1024
#define NHH   16
#define DHH   64
#define HREAL 2730
#define HP    2752          // H padded to multiple of 32
#define MROWS 4096          // B*L

typedef __attribute__((ext_vector_type(16))) __bf16       v16bf;
typedef __attribute__((ext_vector_type(8)))  float        v8f;
typedef __attribute__((ext_vector_type(4)))  unsigned int u32x4;

union AFrag {
  v16bf v;
  unsigned int u[8];
  u32x4 q[2];
};

__device__ __forceinline__ v8f wmma_bf16(v16bf a, v16bf b, v8f c) {
  return __builtin_amdgcn_wmma_f32_16x16x32_bf16(false, a, false, b, (short)0, c,
                                                 false, false);
}

// A fragment: 16x32 bf16, A row-major [*, lda]; per ISA layout lane(l): M=l%16,
// VGPR v holds K = (v/4)*16 + (l/16)*8 + (v%4)*2 + {0,1}  (bf16 pair = 1 dword)
__device__ __forceinline__ v16bf load_a_frag(const __bf16* A, int lda, int row0, int k0) {
  const int lane = threadIdx.x & 31;
  const int m = lane & 15, hf = lane >> 4;
  const __bf16* p = A + (size_t)(row0 + m) * lda + (k0 + hf * 8);
  AFrag f;
#pragma unroll
  for (int vv = 0; vv < 8; ++vv) {
    const int kk = ((vv >> 2) << 4) + ((vv & 3) << 1);
    f.u[vv] = *reinterpret_cast<const unsigned int*>(p + kk);
  }
  return f.v;
}

// B fragment from pre-transposed Bt [N, K] row-major; lane(l): N=l%16,
// element e holds K = k0 + (e/8)*16 + (l/16)*8 + e%8  -> 2 x b128
__device__ __forceinline__ v16bf load_bt_frag(const __bf16* Bt, int ldb, int col0, int k0) {
  const int lane = threadIdx.x & 31;
  const int n = lane & 15, hf = lane >> 4;
  const __bf16* p = Bt + (size_t)(col0 + n) * ldb + (k0 + hf * 8);
  AFrag f;
  f.q[0] = *reinterpret_cast<const u32x4*>(p);
  f.q[1] = *reinterpret_cast<const u32x4*>(p + 16);
  return f.v;
}

// ---- async global->LDS staging of one 64-row x 32-k bf16 A tile ----
// 256 threads x 16B = 4KB = whole tile; one ASYNC instruction per wave.
__device__ __forceinline__ void stage_a_async(const __bf16* A, int lda, int row0blk,
                                              int k0, __bf16* tile) {
  const int t = threadIdx.x;            // 0..255
  const int row = t >> 2, seg = t & 3;  // 64 rows x 4 x 16B segments
  const __bf16* g = A + (size_t)(row0blk + row) * lda + (k0 + seg * 8);
  const unsigned lds = (unsigned)(size_t)(tile + row * 32 + seg * 8);
  asm volatile("global_load_async_to_lds_b128 %0, %1, off"
               :: "v"(lds), "v"(g) : "memory");
}
__device__ __forceinline__ void wait_async_le1() {
  asm volatile("s_wait_asynccnt 0x1" ::: "memory");
}

// ---- weight convert: W [K x N] fp32 -> Bt [Np x Kp] bf16 (zero padded) ----
__global__ void convert_t(const float* __restrict__ W, __bf16* __restrict__ Bt,
                          int K, int N, int Kp, int Np) {
  const size_t total = (size_t)Np * Kp;
  const size_t stride = (size_t)gridDim.x * blockDim.x;
  for (size_t i = (size_t)blockIdx.x * blockDim.x + threadIdx.x; i < total; i += stride) {
    const int n = (int)(i / Kp);
    const int k = (int)(i % Kp);
    float v = 0.f;
    if (n < N && k < K) v = W[(size_t)k * N + n];
    Bt[i] = (__bf16)v;
  }
}

// ---- RMSNorm: one block per row, fp32 in, bf16 out ----
__global__ void rmsnorm_kernel(const float* __restrict__ x, const float* __restrict__ w,
                               __bf16* __restrict__ out) {
  const int row = blockIdx.x;
  const float* xr = x + (size_t)row * DD;
  float s = 0.f;
  for (int i = threadIdx.x; i < DD; i += 256) { const float v = xr[i]; s += v * v; }
#pragma unroll
  for (int offm = 16; offm; offm >>= 1) s += __shfl_xor(s, offm, 32);
  __shared__ float red[8];
  __shared__ float rshared;
  if ((threadIdx.x & 31) == 0) red[threadIdx.x >> 5] = s;
  __syncthreads();
  if (threadIdx.x == 0) {
    float t = 0.f;
    for (int i2 = 0; i2 < 8; ++i2) t += red[i2];
    rshared = rsqrtf(t * (1.f / DD) + 1e-8f);
  }
  __syncthreads();
  const float rr = rshared;
  for (int i = threadIdx.x; i < DD; i += 256)
    out[(size_t)row * DD + i] = (__bf16)(w[i] * xr[i] * rr);
}

// ---- generic GEMM: out_f32 = A@B (+bias) (+resid) ----
// block = 64(M) x 256(N); wave = 32(M) x 64(N); async double-buffered A in LDS
__global__ void gemm_bf16_f32(const __bf16* __restrict__ A, const __bf16* __restrict__ Bt,
                              const float* __restrict__ bias, const float* __restrict__ resid,
                              float* __restrict__ out, int M, int N, int K) {
  __shared__ __align__(16) __bf16 atile[2][64 * 32];
  const int wave = threadIdx.x >> 5;
  const int wm = wave & 1, wn = wave >> 1;
  const int row0blk = blockIdx.y * 64;
  int col0 = blockIdx.x * 256 + wn * 64;
  if (col0 + 64 > N) col0 = N - 64;  // clamp: duplicate tile, identical values
  const int rowA = wm * 32;          // wave rows inside LDS tile
  v8f acc[2][4] = {};
  stage_a_async(A, K, row0blk, 0, atile[0]);
  const int ksteps = K / 32;
  for (int it = 0; it < ksteps; ++it) {
    const int k0 = it * 32;
    const int kn = (it + 1 < ksteps) ? (k0 + 32) : k0;  // clamp last prefetch
    stage_a_async(A, K, row0blk, kn, atile[(it + 1) & 1]);
    wait_async_le1();
    __syncthreads();
    const __bf16* pa = atile[it & 1];
    const v16bf a0 = load_a_frag(pa, 32, rowA, 0);
    const v16bf a1 = load_a_frag(pa, 32, rowA + 16, 0);
#pragma unroll
    for (int t = 0; t < 4; ++t) {
      const v16bf b = load_bt_frag(Bt, K, col0 + 16 * t, k0);
      acc[0][t] = wmma_bf16(a0, b, acc[0][t]);
      acc[1][t] = wmma_bf16(a1, b, acc[1][t]);
    }
    __syncthreads();
  }
  const int lane = threadIdx.x & 31;
  const int nn = lane & 15, hf = lane >> 4;
#pragma unroll
  for (int mi = 0; mi < 2; ++mi)
#pragma unroll
    for (int t = 0; t < 4; ++t)
#pragma unroll
      for (int r = 0; r < 8; ++r) {
        const int row = row0blk + rowA + mi * 16 + r + 8 * hf;
        const int col = col0 + 16 * t + nn;
        float vacc = acc[mi][t][r];
        if (bias) vacc += bias[col];
        const size_t idx = (size_t)row * N + col;
        if (resid) vacc += resid[idx];
        out[idx] = vacc;
      }
}

// ---- QKV GEMM: N=3072, K=1024; scatters q/k per-head, v transposed ----
__global__ void gemm_qkv(const __bf16* __restrict__ A, const __bf16* __restrict__ Bt,
                         const float* __restrict__ bias,
                         __bf16* __restrict__ qb, __bf16* __restrict__ kb,
                         __bf16* __restrict__ vtb) {
  __shared__ __align__(16) __bf16 atile[2][64 * 32];
  const int wave = threadIdx.x >> 5;
  const int wm = wave & 1, wn = wave >> 1;
  const int row0blk = blockIdx.y * 64;
  const int col0 = blockIdx.x * 256 + wn * 64;  // N=3072 exact
  const int rowA = wm * 32;
  v8f acc[2][4] = {};
  stage_a_async(A, DD, row0blk, 0, atile[0]);
  const int ksteps = DD / 32;
  for (int it = 0; it < ksteps; ++it) {
    const int k0 = it * 32;
    const int kn = (it + 1 < ksteps) ? (k0 + 32) : k0;
    stage_a_async(A, DD, row0blk, kn, atile[(it + 1) & 1]);
    wait_async_le1();
    __syncthreads();
    const __bf16* pa = atile[it & 1];
    const v16bf a0 = load_a_frag(pa, 32, rowA, 0);
    const v16bf a1 = load_a_frag(pa, 32, rowA + 16, 0);
#pragma unroll
    for (int t = 0; t < 4; ++t) {
      const v16bf b = load_bt_frag(Bt, DD, col0 + 16 * t, k0);
      acc[0][t] = wmma_bf16(a0, b, acc[0][t]);
      acc[1][t] = wmma_bf16(a1, b, acc[1][t]);
    }
    __syncthreads();
  }
  const int lane = threadIdx.x & 31;
  const int nn = lane & 15, hf = lane >> 4;
#pragma unroll
  for (int mi = 0; mi < 2; ++mi)
#pragma unroll
    for (int t = 0; t < 4; ++t)
#pragma unroll
      for (int r = 0; r < 8; ++r) {
        const int grow = row0blk + rowA + mi * 16 + r + 8 * hf;  // 0..4095
        const int col = col0 + 16 * t + nn;                      // 0..3071
        const float val = acc[mi][t][r] + bias[col];
        const int bb = grow >> 10, ii = grow & 1023;
        const int seg = col >> 10, cc = col & 1023;
        const int hh = cc >> 6, dd2 = cc & 63;
        const int bh = bb * NHH + hh;
        const __bf16 bv = (__bf16)val;
        if (seg == 0)      qb [((size_t)bh * LL + ii) * DHH + dd2] = bv;  // [bh,L,DH]
        else if (seg == 1) kb [((size_t)bh * LL + ii) * DHH + dd2] = bv;  // [bh,L,DH]
        else               vtb[((size_t)bh * DHH + dd2) * LL + ii] = bv;  // [bh,DH,L]
      }
}

// ---- flash attention: wave = 16 query rows, key blocks of 32 ----
__global__ void attn_kernel(const __bf16* __restrict__ qb, const __bf16* __restrict__ kb,
                            const __bf16* __restrict__ vtb, __bf16* __restrict__ headb) {
  __shared__ __bf16 ptile[8][16 * 32];
  const int wave = threadIdx.x >> 5;
  const int lane = threadIdx.x & 31;
  const int nn = lane & 15, hf = lane >> 4;
  const int bh = blockIdx.y;
  const int b = bh >> 4, h = bh & 15;
  const int q0 = blockIdx.x * 128 + wave * 16;
  const __bf16* q  = qb  + (size_t)bh * LL * DHH;
  const __bf16* kk = kb  + (size_t)bh * LL * DHH;
  const __bf16* vt = vtb + (size_t)bh * DHH * LL;
  __bf16* pt = &ptile[wave][0];
  const float scale = 0.03125f;  // D^-0.5 = 1/32 (reference uses d, not dh)

  const v16bf qa0 = load_a_frag(q, DHH, q0, 0);
  const v16bf qa1 = load_a_frag(q, DHH, q0, 32);

  v8f o[4] = {};
  float mrow[8], srow[8];
#pragma unroll
  for (int r = 0; r < 8; ++r) { mrow[r] = -1e30f; srow[r] = 0.f; }

  for (int j0 = 0; j0 <= q0 + 15; j0 += 32) {
    v8f s0 = {}, s1 = {};
    s0 = wmma_bf16(qa0, load_bt_frag(kk, DHH, j0, 0), s0);
    s0 = wmma_bf16(qa1, load_bt_frag(kk, DHH, j0, 32), s0);
    s1 = wmma_bf16(qa0, load_bt_frag(kk, DHH, j0 + 16, 0), s1);
    s1 = wmma_bf16(qa1, load_bt_frag(kk, DHH, j0 + 16, 32), s1);
#pragma unroll
    for (int r = 0; r < 8; ++r) {
      const int qi = q0 + r + 8 * hf;
      const float x0 = s0[r] * scale + ((j0 + nn)      <= qi ? 0.f : -1e9f);
      const float x1 = s1[r] * scale + ((j0 + 16 + nn) <= qi ? 0.f : -1e9f);
      float mx = fmaxf(x0, x1);
#pragma unroll
      for (int offm = 8; offm; offm >>= 1) mx = fmaxf(mx, __shfl_xor(mx, offm, 32));
      const float mnew = fmaxf(mrow[r], mx);
      const float corr = __expf(mrow[r] - mnew);
      mrow[r] = mnew;
      const float p0 = __expf(x0 - mnew);
      const float p1 = __expf(x1 - mnew);
      float ps = p0 + p1;
#pragma unroll
      for (int offm = 8; offm; offm >>= 1) ps += __shfl_xor(ps, offm, 32);
      srow[r] = srow[r] * corr + ps;
#pragma unroll
      for (int t = 0; t < 4; ++t) o[t][r] = o[t][r] * corr;
      pt[(r + 8 * hf) * 32 + nn]      = (__bf16)p0;  // P row-major [16][32] in LDS
      pt[(r + 8 * hf) * 32 + 16 + nn] = (__bf16)p1;
    }
    __builtin_amdgcn_wave_barrier();          // DS ops are in-order per wave
    const v16bf pa = load_a_frag(pt, 32, 0, 0);
#pragma unroll
    for (int t = 0; t < 4; ++t)
      o[t] = wmma_bf16(pa, load_bt_frag(vt, LL, 16 * t, j0), o[t]);
    __builtin_amdgcn_wave_barrier();
  }
#pragma unroll
  for (int t = 0; t < 4; ++t)
#pragma unroll
    for (int r = 0; r < 8; ++r) {
      const int row = b * LL + q0 + r + 8 * hf;
      const int col = h * DHH + 16 * t + nn;
      headb[(size_t)row * DD + col] = (__bf16)(o[t][r] / srow[r]);
    }
}

// ---- fused SwiGLU GEMM: g = silu(A@U) * (A@V), N=HP, K=DD ----
__global__ void gemm_swiglu(const __bf16* __restrict__ A, const __bf16* __restrict__ Ut,
                            const __bf16* __restrict__ Vt, __bf16* __restrict__ g) {
  __shared__ __align__(16) __bf16 atile[2][64 * 32];
  const int wave = threadIdx.x >> 5;
  const int wm = wave & 1, wn = wave >> 1;
  const int row0blk = blockIdx.y * 64;
  int col0 = blockIdx.x * 256 + wn * 64;
  if (col0 + 64 > HP) col0 = HP - 64;  // clamp
  const int rowA = wm * 32;
  v8f accu[2][4] = {}, accv[2][4] = {};
  stage_a_async(A, DD, row0blk, 0, atile[0]);
  const int ksteps = DD / 32;
  for (int it = 0; it < ksteps; ++it) {
    const int k0 = it * 32;
    const int kn = (it + 1 < ksteps) ? (k0 + 32) : k0;
    stage_a_async(A, DD, row0blk, kn, atile[(it + 1) & 1]);
    wait_async_le1();
    __syncthreads();
    const __bf16* pa = atile[it & 1];
    const v16bf a0 = load_a_frag(pa, 32, rowA, 0);
    const v16bf a1 = load_a_frag(pa, 32, rowA + 16, 0);
#pragma unroll
    for (int t = 0; t < 4; ++t) {
      const v16bf bu = load_bt_frag(Ut, DD, col0 + 16 * t, k0);
      accu[0][t] = wmma_bf16(a0, bu, accu[0][t]);
      accu[1][t] = wmma_bf16(a1, bu, accu[1][t]);
      const v16bf bv = load_bt_frag(Vt, DD, col0 + 16 * t, k0);
      accv[0][t] = wmma_bf16(a0, bv, accv[0][t]);
      accv[1][t] = wmma_bf16(a1, bv, accv[1][t]);
    }
    __syncthreads();
  }
  const int lane = threadIdx.x & 31;
  const int nn = lane & 15, hf = lane >> 4;
#pragma unroll
  for (int mi = 0; mi < 2; ++mi)
#pragma unroll
    for (int t = 0; t < 4; ++t)
#pragma unroll
      for (int r = 0; r < 8; ++r) {
        const int row = row0blk + rowA + mi * 16 + r + 8 * hf;
        const int col = col0 + 16 * t + nn;
        const float h1 = accu[mi][t][r], h2 = accv[mi][t][r];
        const float sil = h1 / (1.f + __expf(-h1));
        g[(size_t)row * HP + col] = (__bf16)(sil * h2);
      }
}

extern "C" void kernel_launch(void* const* d_in, const int* in_sizes, int n_in,
                              void* d_out, int out_size, void* d_ws, size_t ws_size,
                              hipStream_t stream) {
  (void)in_sizes; (void)n_in; (void)out_size; (void)ws_size;
  const float* x   = (const float*)d_in[0];
  // d_in[1] = mask m: recomputed analytically on device
  const float* wx  = (const float*)d_in[2];
  const float* bx  = (const float*)d_in[3];
  const float* wo  = (const float*)d_in[4];
  const float* bo  = (const float*)d_in[5];
  const float* mhw = (const float*)d_in[6];
  const float* ffw = (const float*)d_in[7];
  const float* uu  = (const float*)d_in[8];
  const float* vv  = (const float*)d_in[9];
  const float* ww  = (const float*)d_in[10];
  float* out = (float*)d_out;

  char* ws = (char*)d_ws;
  size_t off = 0;
  auto alloc = [&](size_t bytes) -> void* {
    void* p = ws + off;
    off += (bytes + 255) & ~(size_t)255;
    return p;
  };
  __bf16* xn1   = (__bf16*)alloc((size_t)MROWS * DD * 2);
  __bf16* wxT   = (__bf16*)alloc((size_t)3 * DD * DD * 2);
  __bf16* woT   = (__bf16*)alloc((size_t)DD * DD * 2);
  __bf16* uT    = (__bf16*)alloc((size_t)HP * DD * 2);
  __bf16* vT    = (__bf16*)alloc((size_t)HP * DD * 2);
  __bf16* wT    = (__bf16*)alloc((size_t)DD * HP * 2);
  __bf16* qb    = (__bf16*)alloc((size_t)MROWS * DD * 2);
  __bf16* kbuf  = (__bf16*)alloc((size_t)MROWS * DD * 2);
  __bf16* vtb   = (__bf16*)alloc((size_t)MROWS * DD * 2);
  __bf16* headb = (__bf16*)alloc((size_t)MROWS * DD * 2);
  float*  x2    = (float*)alloc((size_t)MROWS * DD * 4);
  __bf16* xn2   = (__bf16*)alloc((size_t)MROWS * DD * 2);
  __bf16* g     = (__bf16*)alloc((size_t)MROWS * HP * 2);

  const dim3 blk(256);

  // weights -> transposed bf16 (padded)
  convert_t<<<1024, blk, 0, stream>>>(wx, wxT, DD, 3 * DD, DD, 3 * DD);
  convert_t<<<512,  blk, 0, stream>>>(wo, woT, DD, DD, DD, DD);
  convert_t<<<1024, blk, 0, stream>>>(uu, uT, DD, HREAL, DD, HP);
  convert_t<<<1024, blk, 0, stream>>>(vv, vT, DD, HREAL, DD, HP);
  convert_t<<<1024, blk, 0, stream>>>(ww, wT, HREAL, DD, HP, DD);

  rmsnorm_kernel<<<MROWS, blk, 0, stream>>>(x, mhw, xn1);
  gemm_qkv<<<dim3(12, 64), blk, 0, stream>>>(xn1, wxT, bx, qb, kbuf, vtb);
  attn_kernel<<<dim3(8, 64), blk, 0, stream>>>(qb, kbuf, vtb, headb);
  // x2 = x + head @ wo + bo
  gemm_bf16_f32<<<dim3(4, 64), blk, 0, stream>>>(headb, woT, bo, x, x2, MROWS, DD, DD);
  rmsnorm_kernel<<<MROWS, blk, 0, stream>>>(x2, ffw, xn2);
  gemm_swiglu<<<dim3(11, 64), blk, 0, stream>>>(xn2, uT, vT, g);
  // out = x2 + g @ w
  gemm_bf16_f32<<<dim3(4, 64), blk, 0, stream>>>(g, wT, nullptr, x2, out, MROWS, DD, HP);
}
